// SotaLSTM_18253611008559
// MI455X (gfx1250) — compile-verified, matrix operand order checked by reading, not run on hardware
//
#include <hip/hip_runtime.h>

#define H 1024
#define HB4 4096
#define BATCH 64
#define TSTEPS 256
#define LN_EPS 1e-5f

typedef __attribute__((ext_vector_type(16))) _Float16 v16h;
typedef __attribute__((ext_vector_type(8)))  _Float16 v8h;
typedef __attribute__((ext_vector_type(8)))  float    v8f;

// WMMA 16x16x32 f16 fragment tile layout (ISA 7.12.2), used for BOTH operands:
//   tile (mt_or_nt, kb): 32 lanes x 16 f16 contiguous (1 KB).
//   lane l -> row/col = tile*16 + (l&15); sel = l>>4
//   element e -> k = kb*32 + (sel ? (e<8 ? e+8 : e+16) : (e<8 ? e : e+8))
__device__ __forceinline__ int frag_k(int sel, int e) {
  return sel ? (e < 8 ? e + 8 : e + 16) : (e < 8 ? e : e + 8);
}
// Inverse mapping: (row,k) -> flat index inside packed fragment array.
__device__ __forceinline__ size_t apack_idx(int row, int k) {
  int mt  = row >> 4;
  int kb  = k >> 5;
  int rem = k & 31;
  int sel = (rem >> 3) & 1;
  int e   = (rem & 7) + ((rem & 16) ? 8 : 0);
  int lane = (row & 15) + (sel << 4);
  return (((size_t)mt * 32 + kb) * 32 + lane) * 16 + e;
}

// ---------------------------------------------------------------------------
// Pack combined weights (H x 4H) into B-fragment tiles.
// col j < 3H -> sum of the 3 H-row blocks of src[:3H,:3H]; j >= 3H -> src[3H:,3H:]
// ---------------------------------------------------------------------------
__global__ void pack_weights_kernel(const float* __restrict__ W,
                                    const float* __restrict__ U,
                                    _Float16* __restrict__ Wpk,
                                    _Float16* __restrict__ Upk) {
  int gid  = blockIdx.x * blockDim.x + threadIdx.x;
  int lane = gid & 31;
  int kb   = (gid >> 5) & 31;
  int nt   = (gid >> 10) & 255;
  int mat  = gid >> 18;                 // 0 = W, 1 = U
  if (mat > 1) return;
  const float* src = mat ? U : W;
  _Float16*    dst = mat ? Upk : Wpk;
  int n   = nt * 16 + (lane & 15);
  int sel = lane >> 4;
  size_t base = ((size_t)(nt * 32 + kb) * 32 + lane) * 16;
#pragma unroll
  for (int e = 0; e < 16; ++e) {
    int k = kb * 32 + frag_k(sel, e);
    float v;
    if (n < 3 * H) {
      v = src[(size_t)k * HB4 + n] + src[(size_t)(H + k) * HB4 + n] +
          src[(size_t)(2 * H + k) * HB4 + n];
    } else {
      v = src[(size_t)(3 * H + k) * HB4 + n];
    }
    dst[base + e] = (_Float16)v;
  }
}

// ---------------------------------------------------------------------------
// Convert x (B*T x H, f32 row-major; row r = b*T+t) directly into packed
// A-fragment tiles (f16). Writes are lane-contiguous 32 B.
// ---------------------------------------------------------------------------
__global__ void pack_x_kernel(const float* __restrict__ x,
                              _Float16* __restrict__ Xpk) {
  int gid  = blockIdx.x * blockDim.x + threadIdx.x;
  int lane = gid & 31;
  int kb   = (gid >> 5) & 31;
  int mt   = gid >> 10;                 // 0 .. 1023 (B*T/16)
  int row  = mt * 16 + (lane & 15);
  int sel  = lane >> 4;
  const float* src = x + (size_t)row * H + kb * 32;
  size_t base = (((size_t)mt * 32 + kb) * 32 + lane) * 16;
#pragma unroll
  for (int e = 0; e < 16; ++e) {
    Xpk[base + e] = (_Float16)src[frag_k(sel, e)];
  }
}

__global__ void init_state_kernel(const float* __restrict__ h0,
                                  const float* __restrict__ c0,
                                  _Float16* __restrict__ h16pk,
                                  float* __restrict__ c) {
  int i = blockIdx.x * blockDim.x + threadIdx.x;   // BATCH*H threads
  int b = i >> 10;
  int k = i & (H - 1);
  h16pk[apack_idx(b, k)] = (_Float16)h0[k];
  c[i] = c0[k];
}

// ---------------------------------------------------------------------------
// WMMA GEMM: C[M x 4096] = Apk (packed A-frag tiles, M x 1024)
//                        * Bpk (packed B-frag tiles, 1024 x 4096).
// One wave -> one 16x16 tile over full K: 32 x v_wmma_f32_16x16x32_f16.
// All operand loads are lane-contiguous 2 x b128; A and B share a 1 KB/kb stride.
// Block = 8 waves = 64(M) x 32(N). Grid = (4096/32, M/64).
// permuteTB: output row r = b*T+t scattered to t*B+b (time-major staging).
// ---------------------------------------------------------------------------
__global__ void __launch_bounds__(256)
wmma_gemm_kernel(const _Float16* __restrict__ Apk,
                 const _Float16* __restrict__ Bpk,
                 float* __restrict__ C,
                 int permuteTB) {
  const int tid   = threadIdx.x;
  const int lane  = tid & 31;
  const int wave  = tid >> 5;
  const int mTile = wave & 3;
  const int nTile = wave >> 2;
  const int m0    = blockIdx.y * 64 + mTile * 16;
  const int mtA   = blockIdx.y * 4 + mTile;
  const int nt    = blockIdx.x * 2 + nTile;
  const _Float16* aptr = Apk + (((size_t)mtA * 32) * 32 + lane) * 16;
  const _Float16* bptr = Bpk + (((size_t)nt  * 32) * 32 + lane) * 16;

  v8f acc = {};
#pragma unroll 4
  for (int kb = 0; kb < 32; ++kb) {
    const _Float16* ak = aptr + (size_t)kb * (32 * 16);
    const _Float16* bk = bptr + (size_t)kb * (32 * 16);
    v8h alo = *(const v8h*)(ak);
    v8h ahi = *(const v8h*)(ak + 8);
    v8h blo = *(const v8h*)(bk);
    v8h bhi = *(const v8h*)(bk + 8);
    if (kb + 1 < 32) {
      __builtin_prefetch(ak + 32 * 16, 0, 1);
      __builtin_prefetch(bk + 32 * 16, 0, 1);
    }
    v16h a, b;
#pragma unroll
    for (int e = 0; e < 8; ++e) {
      a[e] = alo[e]; a[e + 8] = ahi[e];
      b[e] = blo[e]; b[e + 8] = bhi[e];
    }
    acc = __builtin_amdgcn_wmma_f32_16x16x32_f16(false, a, false, b,
                                                 (short)0, acc, false, false);
  }
  // C/D layout: lane<16 -> M=m0+i, lane>=16 -> M=m0+8+i; N = nt*16 + (lane&15)
  const int sel  = lane >> 4;
  const int ncol = nt * 16 + (lane & 15);
#pragma unroll
  for (int i = 0; i < 8; ++i) {
    int m    = m0 + sel * 8 + i;
    int outr = permuteTB ? ((m & (TSTEPS - 1)) * BATCH + (m >> 8)) : m;
    C[(size_t)outr * HB4 + ncol] = acc[i];
  }
}

// ---------------------------------------------------------------------------
// Fused per-step elementwise: gates = XW[t] + hU + bias + peephole,
// per-gate LayerNorm, sigmoid/tanh, c/h update; h stored both as f32 output
// and directly into packed A-fragment f16 for the next step's GEMM.
// One block per batch row b, 256 threads.
// ---------------------------------------------------------------------------
__global__ void __launch_bounds__(256)
lstm_step_kernel(const float* __restrict__ XWt,   // BATCH x 4096 slice for t
                 const float* __restrict__ hU,    // BATCH x 4096
                 const float* __restrict__ bias,  // 4096
                 const float* __restrict__ wp,    // 3072
                 const float* __restrict__ g_i, const float* __restrict__ be_i,
                 const float* __restrict__ g_f, const float* __restrict__ be_f,
                 const float* __restrict__ g_o, const float* __restrict__ be_o,
                 const float* __restrict__ g_c, const float* __restrict__ be_c,
                 float* __restrict__ c,           // BATCH x H
                 _Float16* __restrict__ h16pk,    // packed A-frag, BATCH x H
                 float* __restrict__ out,         // B x T x H
                 int t) {
  __shared__ float  zb[4][H];
  __shared__ float2 rbuf[256];
  const int b   = blockIdx.x;
  const int tid = threadIdx.x;
  const float* xw   = XWt + (size_t)b * HB4;
  const float* hu   = hU  + (size_t)b * HB4;
  float*       crow = c   + (size_t)b * H;

  float mean[4], rstd[4];
  for (int g = 0; g < 4; ++g) {
    float s = 0.f, ss = 0.f;
    for (int k = tid; k < H; k += 256) {
      float z = xw[g * H + k] + hu[g * H + k] + bias[g * H + k];
      if (g < 3) z += wp[g * H + k] * crow[k];
      zb[g][k] = z;
      s += z;
      ss += z * z;
    }
    rbuf[tid] = make_float2(s, ss);
    __syncthreads();
    for (int o = 128; o > 0; o >>= 1) {
      if (tid < o) {
        rbuf[tid].x += rbuf[tid + o].x;
        rbuf[tid].y += rbuf[tid + o].y;
      }
      __syncthreads();
    }
    float m   = rbuf[0].x * (1.f / H);
    float var = rbuf[0].y * (1.f / H) - m * m;
    mean[g] = m;
    rstd[g] = rsqrtf(var + LN_EPS);
    __syncthreads();
  }

  for (int k = tid; k < H; k += 256) {
    float iv = (zb[0][k] - mean[0]) * rstd[0] * g_i[k] + be_i[k];
    float fv = (zb[1][k] - mean[1]) * rstd[1] * g_f[k] + be_f[k];
    float ov = (zb[2][k] - mean[2]) * rstd[2] * g_o[k] + be_o[k];
    float gc = (zb[3][k] - mean[3]) * rstd[3] * g_c[k] + be_c[k];
    float i_t = 1.f / (1.f + expf(-iv));
    float f_t = 1.f / (1.f + expf(-fv));
    float o_t = 1.f / (1.f + expf(-ov));
    float cn  = f_t * crow[k] + i_t * tanhf(gc);
    float hn  = tanhf(cn) * o_t;
    crow[k] = cn;
    h16pk[apack_idx(b, k)] = (_Float16)hn;
    out[((size_t)b * TSTEPS + t) * H + k] = hn;
  }
}

extern "C" void kernel_launch(void* const* d_in, const int* in_sizes, int n_in,
                              void* d_out, int out_size, void* d_ws, size_t ws_size,
                              hipStream_t stream) {
  const float* x    = (const float*)d_in[0];
  const float* W    = (const float*)d_in[1];
  const float* U    = (const float*)d_in[2];
  const float* bv   = (const float*)d_in[3];
  const float* wp   = (const float*)d_in[4];
  const float* h0   = (const float*)d_in[5];
  const float* c0   = (const float*)d_in[6];
  const float* g_i  = (const float*)d_in[7];
  const float* be_i = (const float*)d_in[8];
  const float* g_f  = (const float*)d_in[9];
  const float* be_f = (const float*)d_in[10];
  const float* g_o  = (const float*)d_in[11];
  const float* be_o = (const float*)d_in[12];
  const float* g_c  = (const float*)d_in[13];
  const float* be_c = (const float*)d_in[14];
  float* out = (float*)d_out;

  char* ws = (char*)d_ws;
  size_t off = 0;
  auto carve = [&](size_t bytes) -> char* {
    char* p = ws + off;
    off += (bytes + 255) & ~(size_t)255;
    return p;
  };
  _Float16* Wpk   = (_Float16*)carve((size_t)H * HB4 * sizeof(_Float16));
  _Float16* Upk   = (_Float16*)carve((size_t)H * HB4 * sizeof(_Float16));
  _Float16* Xpk   = (_Float16*)carve((size_t)BATCH * TSTEPS * H * sizeof(_Float16));
  float*    XW    = (float*)carve((size_t)BATCH * TSTEPS * HB4 * sizeof(float));
  float*    hU    = (float*)carve((size_t)BATCH * HB4 * sizeof(float));
  float*    cbuf  = (float*)carve((size_t)BATCH * H * sizeof(float));
  _Float16* h16pk = (_Float16*)carve((size_t)BATCH * H * sizeof(_Float16));

  // One-time (per launch) preprocessing — fully parallel.
  pack_weights_kernel<<<2048, 256, 0, stream>>>(W, U, Wpk, Upk);
  pack_x_kernel<<<(BATCH * TSTEPS / 16) * 32 * 32 / 256, 256, 0, stream>>>(x, Xpk);
  init_state_kernel<<<BATCH * H / 256, 256, 0, stream>>>(h0, c0, h16pk, cbuf);

  // Bulk input projection for all timesteps: (B*T x H) @ (H x 4H), time-major out.
  dim3 gx(HB4 / 32, (BATCH * TSTEPS) / 64);
  wmma_gemm_kernel<<<gx, 256, 0, stream>>>(Xpk, Wpk, XW, 1);

  // Sequential recurrence: tiny WMMA GEMM + fused LN/activation per step.
  dim3 gh(HB4 / 32, 1);
  for (int t = 0; t < TSTEPS; ++t) {
    wmma_gemm_kernel<<<gh, 256, 0, stream>>>(h16pk, Upk, hU, 0);
    lstm_step_kernel<<<BATCH, 256, 0, stream>>>(
        XW + (size_t)t * BATCH * HB4, hU, bv, wp,
        g_i, be_i, g_f, be_f, g_o, be_o, g_c, be_c,
        cbuf, h16pk, out, t);
  }
  (void)in_sizes; (void)n_in; (void)out_size; (void)ws_size;
}
